// GRUnet_88837103551249
// MI455X (gfx1250) — matrix operand order repaired and checked by reading, not compiled, not run on hardware
//
#include <hip/hip_runtime.h>

constexpr int kH = 10;   // hidden size
constexpr int kI = 3;    // input size
constexpr int kT = 1000; // timesteps
constexpr int kB = 4096; // batch
constexpr int kWavesPerBlock = 8;
constexpr int kRowsPerWave = 16;

typedef __attribute__((ext_vector_type(16))) _Float16 v16h;
typedef __attribute__((ext_vector_type(8)))  float    v8f;
typedef __attribute__((ext_vector_type(8)))  unsigned v8u;

__device__ __forceinline__ float fast_tanh(float x) {
#if __has_builtin(__builtin_amdgcn_tanhf)
  return __builtin_amdgcn_tanhf(x);
#else
  float y;
  asm("v_tanh_f32 %0, %1" : "=v"(y) : "v"(x));
  return y;
#endif
}

__device__ __forceinline__ float fast_sigmoid(float x) {
  // sigmoid(x) = 0.5*tanh(0.5x) + 0.5  (exact identity)
  return __builtin_fmaf(0.5f, fast_tanh(0.5f * x), 0.5f);
}

__device__ __forceinline__ unsigned pk16(float a, float b) {
#if __has_builtin(__builtin_amdgcn_cvt_pkrtz)
  auto h = __builtin_amdgcn_cvt_pkrtz(a, b);  // __fp16 ext_vector_type(2)
  return __builtin_bit_cast(unsigned, h);
#else
  unsigned lo16 = (unsigned)__builtin_bit_cast(unsigned short, (_Float16)a);
  unsigned hi16 = (unsigned)__builtin_bit_cast(unsigned short, (_Float16)b);
  return lo16 | (hi16 << 16);
#endif
}

// Exchange lane L <-> lane L^16 (ds_swizzle group-of-32, xor=0x10, and=0x1f).
__device__ __forceinline__ unsigned swap16(unsigned x) {
  return (unsigned)__builtin_amdgcn_ds_swizzle((int)x, 0x401F);
}

// Build a 16x32 f16 WMMA A-matrix tile for one gate.
// Row m = hidden unit of the gate (rows >= kH are zero).
// K layout: K=0..in_dim-1 -> Wih row, K=15 -> folded bias (paired with a
// constant 1.0 at K=15 in the B operand), K=16..16+kH-1 -> Whh row.
// 16-bit A layout (contiguous): lanes 0-15 vgpr v -> K=2v,2v+1;
// lanes 16-31 vgpr v -> K=16+2v,16+2v+1.
__device__ v16h build_tile(const float* Wih, int in_dim, const float* Whh,
                           const float* b_ih, const float* b_hh,
                           int gate, int lane) {
  const int m = lane & 15;
  const int hi = lane >> 4;
  v16h a = {};
  if (m < kH) {
    const int row = gate * kH + m;
#pragma unroll
    for (int v = 0; v < 8; ++v) {
#pragma unroll
      for (int e = 0; e < 2; ++e) {
        int k = hi * 16 + 2 * v + e;
        float w = 0.0f;
        if (k == 15) {
          if (b_ih) w += b_ih[row];
          if (b_hh) w += b_hh[row];
        } else if (k < 16) {
          if (Wih && k < in_dim) w = Wih[row * in_dim + k];
        } else {
          int kk = k - 16;
          if (Whh && kk < kH) w = Whh[row * kH + kk];
        }
        a[2 * v + e] = (_Float16)w;
      }
    }
  }
  return a;
}

struct LayerState {
  v8f H;          // h in f32 C-tile layout: lane n vgpr v = h[n][v] (lanes 0-15),
                  // lane n+16 vgpr v = h[n][8+v] (only v<2 meaningful)
  unsigned P[5];  // h packed f16 pairs, valid in UPPER lanes: recurrent B half
};

// One GRU layer step for a 16-row batch tile.
// Xin: layer input packed f16 pairs (valid in lower lanes), K=0..9 (+zeros).
// Produces Xout (this layer's h packed for the next layer's input) and
// updates st.H / st.P.
__device__ __forceinline__ void gru_step(const v16h& Ar, const v16h& Az,
                                         const v16h& Axn, const v16h& Ahn,
                                         const unsigned* Xin,
                                         LayerState& st, unsigned* Xout,
                                         bool lo) {
  // Assemble shared B operand: lower lanes K=0..15 = input (+1.0 at K=15),
  // upper lanes K=16..31 = recurrent h.
  v8u bu;
#pragma unroll
  for (int v = 0; v < 5; ++v) bu[v] = lo ? Xin[v] : st.P[v];
  bu[5] = 0u;
  bu[6] = 0u;
  bu[7] = lo ? 0x3C000000u : 0u;  // K=15 = 1.0h (bias column), K=14/30/31 = 0
  v16h B = __builtin_bit_cast(v16h, bu);

  const v8f c0 = {};
  v8f aR  = __builtin_amdgcn_wmma_f32_16x16x32_f16(false, Ar,  false, B, (short)0, c0, false, false);
  v8f aZ  = __builtin_amdgcn_wmma_f32_16x16x32_f16(false, Az,  false, B, (short)0, c0, false, false);
  v8f aXN = __builtin_amdgcn_wmma_f32_16x16x32_f16(false, Axn, false, B, (short)0, c0, false, false);
  v8f aHN = __builtin_amdgcn_wmma_f32_16x16x32_f16(false, Ahn, false, B, (short)0, c0, false, false);

  // Elementwise GRU update entirely in the C-tile layout.
#pragma unroll
  for (int i = 0; i < 8; ++i) {
    float r = fast_sigmoid(aR[i]);
    float z = fast_sigmoid(aZ[i]);
    float n = fast_tanh(__builtin_fmaf(r, aHN[i], aXN[i]));
    st.H[i] = __builtin_fmaf(z, st.H[i] - n, n);  // (1-z)*n + z*h
  }

  // Repack h for the B operand of (a) next layer's input half and
  // (b) this layer's recurrent half next step.
  unsigned p0 = pk16(st.H[0], st.H[1]);  // lower: (h0,h1)  upper: (h8,h9)
  unsigned p1 = pk16(st.H[2], st.H[3]);
  unsigned p2 = pk16(st.H[4], st.H[5]);
  unsigned p3 = pk16(st.H[6], st.H[7]);
  unsigned s0 = swap16(p0);              // lower: (h8,h9)  upper: (h0,h1)
  Xout[0] = p0; Xout[1] = p1; Xout[2] = p2; Xout[3] = p3; Xout[4] = s0;
  st.P[0] = s0;
  st.P[1] = swap16(p1);
  st.P[2] = swap16(p2);
  st.P[3] = swap16(p3);
  st.P[4] = p0;
}

__global__ __launch_bounds__(kWavesPerBlock * 32) void gru3_wmma_kernel(
    const float* __restrict__ x,
    const float* __restrict__ Wih0, const float* __restrict__ Whh0,
    const float* __restrict__ bih0, const float* __restrict__ bhh0,
    const float* __restrict__ Wih1, const float* __restrict__ Whh1,
    const float* __restrict__ bih1, const float* __restrict__ bhh1,
    const float* __restrict__ Wih2, const float* __restrict__ Whh2,
    const float* __restrict__ bih2, const float* __restrict__ bhh2,
    const float* __restrict__ Wout, const float* __restrict__ bout,
    float* __restrict__ out) {
  const int lane = (int)(threadIdx.x & 31u);
  const int wave = (int)(threadIdx.x >> 5u);
  const int b0 = ((int)blockIdx.x * kWavesPerBlock + wave) * kRowsPerWave;
  const bool lo = lane < 16;

  // Preload all weight tiles into VGPRs (12 tiles x 8 VGPRs = 96 VGPRs).
  v16h Ar0  = build_tile(Wih0, kI, Whh0, bih0, bhh0, 0, lane);
  v16h Az0  = build_tile(Wih0, kI, Whh0, bih0, bhh0, 1, lane);
  v16h Axn0 = build_tile(Wih0, kI, nullptr, bih0, nullptr, 2, lane);
  v16h Ahn0 = build_tile(nullptr, 0, Whh0, nullptr, bhh0, 2, lane);

  v16h Ar1  = build_tile(Wih1, kH, Whh1, bih1, bhh1, 0, lane);
  v16h Az1  = build_tile(Wih1, kH, Whh1, bih1, bhh1, 1, lane);
  v16h Axn1 = build_tile(Wih1, kH, nullptr, bih1, nullptr, 2, lane);
  v16h Ahn1 = build_tile(nullptr, 0, Whh1, nullptr, bhh1, 2, lane);

  v16h Ar2  = build_tile(Wih2, kH, Whh2, bih2, bhh2, 0, lane);
  v16h Az2  = build_tile(Wih2, kH, Whh2, bih2, bhh2, 1, lane);
  v16h Axn2 = build_tile(Wih2, kH, nullptr, bih2, nullptr, 2, lane);
  v16h Ahn2 = build_tile(nullptr, 0, Whh2, nullptr, bhh2, 2, lane);

  LayerState s0 = {}, s1 = {}, s2 = {};  // h0 = 0

  const float* xp = x + (size_t)(b0 + (lane & 15)) * (size_t)(kT * kI);

  for (int t = 0; t < kT; ++t) {
    float x0 = xp[3 * t + 0];
    float x1 = xp[3 * t + 1];
    float x2 = xp[3 * t + 2];
    unsigned X0[5] = {pk16(x0, x1), pk16(x2, 0.0f), 0u, 0u, 0u};
    unsigned X1[5], X2[5], X3[5];
    gru_step(Ar0, Az0, Axn0, Ahn0, X0, s0, X1, lo);
    gru_step(Ar1, Az1, Axn1, Ahn1, X1, s1, X2, lo);
    gru_step(Ar2, Az2, Axn2, Ahn2, X2, s2, X3, lo);
  }

  // out[b] = dot(h2[b], Wout) + bout.  Lower lanes hold hid 0..7, upper hold
  // hid 8..9; reduce across lane halves with one swizzle.
  float partial = 0.0f;
#pragma unroll
  for (int v = 0; v < 8; ++v) {
    int j = (lo ? 0 : 8) + v;
    float w = (j < kH) ? Wout[j] : 0.0f;
    partial = __builtin_fmaf(w, s2.H[v], partial);
  }
  float other = __builtin_bit_cast(float, swap16(__builtin_bit_cast(unsigned, partial)));
  if (lo) out[b0 + lane] = partial + other + bout[0];
}

extern "C" void kernel_launch(void* const* d_in, const int* /*in_sizes*/, int /*n_in*/,
                              void* d_out, int /*out_size*/, void* /*d_ws*/, size_t /*ws_size*/,
                              hipStream_t stream) {
  const float* x    = (const float*)d_in[0];
  const float* Wih0 = (const float*)d_in[1];
  const float* Whh0 = (const float*)d_in[2];
  const float* bih0 = (const float*)d_in[3];
  const float* bhh0 = (const float*)d_in[4];
  const float* Wih1 = (const float*)d_in[5];
  const float* Whh1 = (const float*)d_in[6];
  const float* bih1 = (const float*)d_in[7];
  const float* bhh1 = (const float*)d_in[8];
  const float* Wih2 = (const float*)d_in[9];
  const float* Whh2 = (const float*)d_in[10];
  const float* bih2 = (const float*)d_in[11];
  const float* bhh2 = (const float*)d_in[12];
  const float* Wout = (const float*)d_in[13];
  const float* bout = (const float*)d_in[14];

  const int blocks = kB / (kWavesPerBlock * kRowsPerWave);  // 32 blocks, 256 waves
  gru3_wmma_kernel<<<blocks, kWavesPerBlock * 32, 0, stream>>>(
      x, Wih0, Whh0, bih0, bhh0, Wih1, Whh1, bih1, bhh1,
      Wih2, Whh2, bih2, bhh2, Wout, bout, (float*)d_out);
}
